// VQVAE_GCN_76261439307888
// MI455X (gfx1250) — compile-verified, matrix operand order checked by reading, not run on hardware
//
#include <hip/hip_runtime.h>
#include <hip/hip_bf16.h>
#include <math.h>

// ---------------- types ----------------
typedef __bf16 bf16;
typedef __attribute__((ext_vector_type(16))) __bf16 bf16x16;
typedef __attribute__((ext_vector_type(8)))  __bf16 bf16x8;
typedef __attribute__((ext_vector_type(8)))  float  f32x8;

__device__ __forceinline__ bf16 f2bf(float f) {
  union { float f; unsigned u; } v; v.f = f;
  unsigned r = v.u + 0x7FFFu + ((v.u >> 16) & 1u);   // round-to-nearest-even
  unsigned short h = (unsigned short)(r >> 16);
  bf16 out; __builtin_memcpy(&out, &h, 2);
  return out;
}

// LDS byte offset of a generic pointer that points into __shared__ space
__device__ __forceinline__ unsigned lds_off(void* p) {
  return (unsigned)(unsigned long long)(__attribute__((address_space(3))) void*)p;
}

// CDNA5 async global->LDS 16B copy (per-lane), tracked by ASYNCcnt
__device__ __forceinline__ void async_g2l_b128(const void* g, void* l) {
  unsigned long long ga = (unsigned long long)g;
  unsigned lo = lds_off(l);
  asm volatile("global_load_async_to_lds_b128 %0, %1, off"
               :: "v"(lo), "v"(ga) : "memory");
}
__device__ __forceinline__ void wait_async0() {
  asm volatile("s_wait_asynccnt 0x0" ::: "memory");
}

// ---------------- problem constants ----------------
#define NB    8
#define HDIM  128
#define RESH  64
#define NE    1024
#define EDIM  64
#define NPIX  8192            // 8*32*32 latent pixels
#define K_C2  1024            // 64ch * 4*4
#define K_R1  1152            // 128ch * 3*3

// output layout (floats): loss | z_q[8,64,32,32] | perplexity | emb[1024,64] | idx[8192]
#define OUT_LOSS 0
#define OUT_ZQ   1
#define OUT_PERP (1 + NB*EDIM*32*32)                 // 524289
#define OUT_EMB  (OUT_PERP + 1)                      // 524290
#define OUT_IDX  (OUT_EMB + NE*EDIM)                 // 589826

// =====================================================================
// bf16 WMMA GEMM:  C[M,N] (+)= A[M,K] * Bt[N,K]^T  (+bias)(relu)
// Block = 256 threads = 8 waves; macro tile 128(M) x 64(N).
//   - B panels double-buffered in LDS via async global->LDS DMA
//     (stage chunk i+1 overlaps the 8 WMMAs of chunk i; 1 barrier/chunk)
//   - each wave computes 16x64 (4 accumulator tiles), A fragment reused 4x
// Requires: M % 128 == 0, N % 64 == 0, K % 64 == 0 (true for all calls).
// flags: bit0 = accumulate into existing C, bit1 = relu epilogue
// =====================================================================
#define KC 64
__global__ __launch_bounds__(256)
void gemm_bf16_wmma(const bf16* __restrict__ A, const bf16* __restrict__ Bt,
                    float* __restrict__ C, const float* __restrict__ bias,
                    int M, int N, int K, int flags) {
  __shared__ __align__(16) bf16 Bs[2][64 * KC];       // 2 x 8KB B panels
  const int wave = threadIdx.x >> 5;
  const int lane = threadIdx.x & 31;
  const int nBlkN = N >> 6;
  const int bM = blockIdx.x / nBlkN;
  const int bN = blockIdx.x % nBlkN;
  const int laneLow  = lane & 15;
  const int laneHalf = lane >> 4;

  const int mRow = bM * 128 + wave * 16 + laneLow;
  const bf16* aRow = A + (size_t)mRow * K + laneHalf * 8;

  // per-thread B staging coords: 512 x 16B chunks per panel, 2 per thread
  const int c0 = threadIdx.x, c1 = threadIdx.x + 256;
  const int sn0 = c0 >> 3, sq0 = c0 & 7;
  const int sn1 = c1 >> 3, sq1 = c1 & 7;
  const bf16* bPan = Bt + (size_t)(bN * 64) * K;

  f32x8 acc[4] = {{}, {}, {}, {}};
  const int nCh = K / KC;

  // prologue: stage chunk 0 into buffer 0
  async_g2l_b128(bPan + (size_t)sn0 * K + 0 * KC + sq0 * 8, &Bs[0][sn0 * KC + sq0 * 8]);
  async_g2l_b128(bPan + (size_t)sn1 * K + 0 * KC + sq1 * 8, &Bs[0][sn1 * KC + sq1 * 8]);

  for (int i = 0; i < nCh; ++i) {
    wait_async0();                                    // my panel-i DMAs done
    __syncthreads();                                  // everyone's done; prev reads done
    const int buf = i & 1;
    if (i + 1 < nCh) {                                // overlap next DMA with compute
      const int kn = (i + 1) * KC;
      async_g2l_b128(bPan + (size_t)sn0 * K + kn + sq0 * 8, &Bs[buf ^ 1][sn0 * KC + sq0 * 8]);
      async_g2l_b128(bPan + (size_t)sn1 * K + kn + sq1 * 8, &Bs[buf ^ 1][sn1 * KC + sq1 * 8]);
      __builtin_prefetch(aRow + (i + 1) * KC, 0, 0);  // next A chunk -> cache
    }
    const int kk = i * KC;
#pragma unroll
    for (int ks = 0; ks < KC; ks += 32) {
      bf16x8 a0 = *(const bf16x8*)(aRow + kk + ks);
      bf16x8 a1 = *(const bf16x8*)(aRow + kk + ks + 16);
      bf16x16 av = __builtin_shufflevector(a0, a1,
          0,1,2,3,4,5,6,7,8,9,10,11,12,13,14,15);
#pragma unroll
      for (int j = 0; j < 4; ++j) {                   // reuse A across 4 N-tiles
        const bf16* bp = &Bs[buf][(j * 16 + laneLow) * KC + ks + laneHalf * 8];
        bf16x8 b0 = *(const bf16x8*)bp;               // ds_load_b128
        bf16x8 b1 = *(const bf16x8*)(bp + 16);
        bf16x16 bv = __builtin_shufflevector(b0, b1,
            0,1,2,3,4,5,6,7,8,9,10,11,12,13,14,15);
        acc[j] = __builtin_amdgcn_wmma_f32_16x16x32_bf16(
            false, av, false, bv, (short)0, acc[j], false, false);
      }
    }
  }

  // ---- epilogue ----
#pragma unroll
  for (int j = 0; j < 4; ++j) {
    const int col = bN * 64 + j * 16 + laneLow;
    const float bb = bias ? bias[col] : 0.0f;
#pragma unroll
    for (int v = 0; v < 8; ++v) {
      const int row = bM * 128 + wave * 16 + v + laneHalf * 8;
      const size_t off = (size_t)row * N + col;
      float r = acc[j][v] + bb;
      if (flags & 1) r += C[off];
      if (flags & 2) r = fmaxf(r, 0.0f);
      C[off] = r;
    }
  }
}

// =====================================================================
// conv1: x[8,3,512,512] f32 -> h1[8,128,128,64] NHWC bf16, stride4 4x4, relu
// =====================================================================
__global__ void k_conv1(const float* __restrict__ x, const float* __restrict__ w,
                        const float* __restrict__ bia, bf16* __restrict__ h1) {
  int t = blockIdx.x * blockDim.x + threadIdx.x;
  if (t >= NB * 128 * 128 * 64) return;
  int oc = t & 63;
  int p  = t >> 6;
  int ox = p & 127; p >>= 7;
  int oy = p & 127; p >>= 7;
  int b  = p;
  float s = bia[oc];
  const float* xb = x + (size_t)b * 3 * 512 * 512;
#pragma unroll
  for (int c = 0; c < 3; ++c)
#pragma unroll
    for (int ky = 0; ky < 4; ++ky)
#pragma unroll
      for (int kx = 0; kx < 4; ++kx)
        s += xb[(size_t)c * 512 * 512 + (size_t)(oy * 4 + ky) * 512 + (ox * 4 + kx)]
           * w[((oc * 3 + c) * 4 + ky) * 4 + kx];
  h1[t] = f2bf(fmaxf(s, 0.0f));
}

// =====================================================================
// pack OIHW f32 weights -> [O][(ky*KW+kx)*Ci + i] bf16  (K-contig per out ch)
// =====================================================================
__global__ void k_pack_w(const float* __restrict__ src, bf16* __restrict__ dst,
                         int O, int Ci, int KH, int KW) {
  int t = blockIdx.x * blockDim.x + threadIdx.x;
  if (t >= O * Ci * KH * KW) return;
  int kx = t % KW; int r = t / KW;
  int ky = r % KH; r /= KH;
  int i  = r % Ci; int o = r / Ci;
  dst[(size_t)o * (Ci * KH * KW) + (size_t)(ky * KW + kx) * Ci + i] = f2bf(src[t]);
}

// =====================================================================
// im2col for conv2: h1 NHWC [8,128,128,64] -> A2[8192,1024]
// =====================================================================
__global__ void k_im2col_c2(const bf16* __restrict__ h1, bf16* __restrict__ A2) {
  int t = blockIdx.x * blockDim.x + threadIdx.x;    // NPIX*16*8
  if (t >= NPIX * 16 * 8) return;
  int ch8 = t & 7;
  int s   = (t >> 3) & 15;
  int p   = t >> 7;
  int ox = p & 31, oy = (p >> 5) & 31, b = p >> 10;
  int ky = s >> 2, kx = s & 3;
  const bf16x8* src = (const bf16x8*)(h1 +
      ((size_t)((b * 128 + oy * 4 + ky) * 128 + ox * 4 + kx)) * 64 + ch8 * 8);
  *(bf16x8*)(A2 + (size_t)p * K_C2 + s * 64 + ch8 * 8) = *src;
}

// =====================================================================
// im2col 3x3 SAME: hb [8192,128] bf16 NHWC(32x32) -> A3[8192,1152]
// =====================================================================
__global__ void k_im2col_r(const bf16* __restrict__ hb, bf16* __restrict__ A3) {
  int t = blockIdx.x * blockDim.x + threadIdx.x;    // NPIX*9*16
  if (t >= NPIX * 9 * 16) return;
  int ch8 = t % 16; int r = t / 16;
  int s   = r % 9;  int p = r / 9;
  int x = p & 31, y = (p >> 5) & 31, b = p >> 10;
  int ky = s / 3, kx = s % 3;
  int yy = y + ky - 1, xx = x + kx - 1;
  bf16x8 v = {};
  if (yy >= 0 && yy < 32 && xx >= 0 && xx < 32)
    v = *(const bf16x8*)(hb + ((size_t)((b * 32 + yy) * 32 + xx)) * 128 + ch8 * 8);
  *(bf16x8*)(A3 + (size_t)p * K_R1 + s * 128 + ch8 * 8) = v;
}

// =====================================================================
__global__ void k_relu_cast(const float* __restrict__ src, bf16* __restrict__ dst,
                            int n, int doRelu) {
  int t = blockIdx.x * blockDim.x + threadIdx.x;
  if (t >= n) return;
  float v = src[t];
  if (doRelu) v = fmaxf(v, 0.0f);
  dst[t] = f2bf(v);
}

__global__ void k_enorm(const float* __restrict__ emb, float* __restrict__ en) {
  int n = blockIdx.x * blockDim.x + threadIdx.x;
  if (n >= NE) return;
  float s = 0.0f;
  for (int c = 0; c < EDIM; ++c) { float e = emb[(size_t)n * EDIM + c]; s += e * e; }
  en[n] = s;
}

__global__ void k_init(int* __restrict__ counts, float* __restrict__ mse) {
  int t = blockIdx.x * blockDim.x + threadIdx.x;
  if (t < NE) counts[t] = 0;
  if (t == NE) *mse = 0.0f;
}

// =====================================================================
// argmin per latent pixel over 1024 codes: one wave32 per row
// d(n) = |e_n|^2 - 2 * dot(z, e_n)   (|z|^2 constant per row)
// =====================================================================
__global__ __launch_bounds__(256)
void k_argmin(const float* __restrict__ dots, const float* __restrict__ en,
              int* __restrict__ idx, int* __restrict__ counts) {
  int m = blockIdx.x * 8 + (threadIdx.x >> 5);
  int lane = threadIdx.x & 31;
  const float* dr = dots + (size_t)m * NE;
  float best = 3.4e38f; int bi = 0x7FFFFFFF;
  for (int n = lane; n < NE; n += 32) {
    float d = en[n] - 2.0f * dr[n];
    if (d < best) { best = d; bi = n; }
  }
  for (int off = 16; off > 0; off >>= 1) {
    float ob = __shfl_xor(best, off, 32);
    int   oi = __shfl_xor(bi,   off, 32);
    if (ob < best || (ob == best && oi < bi)) { best = ob; bi = oi; }
  }
  if (lane == 0) { idx[m] = bi; atomicAdd(&counts[bi], 1); }
}

// =====================================================================
// z_q output (NCHW) + MSE accumulation. Forward z_q == quantized latents.
// =====================================================================
__global__ __launch_bounds__(256)
void k_zq(const int* __restrict__ idx, const float* __restrict__ emb,
          const float* __restrict__ ze, float* __restrict__ out,
          float* __restrict__ mseAcc) {
  int t = blockIdx.x * blockDim.x + threadIdx.x;    // NPIX*EDIM
  int c = t & 63, m = t >> 6;
  int x = m & 31, y = (m >> 5) & 31, b = m >> 10;
  float q = emb[(size_t)idx[m] * EDIM + c];
  float d = q - ze[t];
  out[OUT_ZQ + (((size_t)b * EDIM + c) * 32 + y) * 32 + x] = q;
  float s = d * d;
  for (int off = 16; off > 0; off >>= 1) s += __shfl_xor(s, off, 32);
  if ((threadIdx.x & 31) == 0) atomicAdd(mseAcc, s);
}

// loss = (1+beta)*mse ; perplexity = exp(-sum p log(p+1e-10))
__global__ void k_scalars(const int* __restrict__ counts,
                          const float* __restrict__ mseAcc, float* __restrict__ out) {
  __shared__ float sh[32];
  int t = threadIdx.x;                               // 1024 threads
  float p = (float)counts[t] * (1.0f / (float)NPIX);
  float s = p * logf(p + 1e-10f);
  for (int off = 16; off > 0; off >>= 1) s += __shfl_xor(s, off, 32);
  if ((t & 31) == 0) sh[t >> 5] = s;
  __syncthreads();
  if (t < 32) {
    float v = sh[t];
    for (int off = 16; off > 0; off >>= 1) v += __shfl_xor(v, off, 32);
    if (t == 0) {
      out[OUT_PERP] = expf(-v);
      out[OUT_LOSS] = 1.25f * (*mseAcc) / (float)(NPIX * EDIM);
    }
  }
}

__global__ void k_copyout(const float* __restrict__ emb, const int* __restrict__ idx,
                          float* __restrict__ out) {
  int t = blockIdx.x * blockDim.x + threadIdx.x;
  if (t < NE * EDIM)                out[OUT_EMB + t] = emb[t];
  else if (t < NE * EDIM + NPIX)    out[OUT_IDX + (t - NE * EDIM)] = (float)idx[t - NE * EDIM];
}

// =====================================================================
extern "C" void kernel_launch(void* const* d_in, const int* in_sizes, int n_in,
                              void* d_out, int out_size, void* d_ws, size_t ws_size,
                              hipStream_t stream) {
  const float* x       = (const float*)d_in[0];
  const float* conv1_w = (const float*)d_in[1];
  const float* conv1_b = (const float*)d_in[2];
  const float* conv2_w = (const float*)d_in[3];
  const float* conv2_b = (const float*)d_in[4];
  const float* res_w1  = (const float*)d_in[5];
  const float* res_w2  = (const float*)d_in[6];
  const float* preq_w  = (const float*)d_in[7];
  const float* preq_b  = (const float*)d_in[8];
  const float* emb     = (const float*)d_in[9];
  float* out = (float*)d_out;

  // ---- workspace carve-up (all offsets 256B aligned) ----
  char*  base = (char*)d_ws;
  size_t o = 0;
  auto carve = [&](size_t bytes) { char* p = base + o; o = (o + bytes + 255) & ~(size_t)255; return p; };
  bf16* h1   = (bf16*)carve((size_t)NB * 128 * 128 * 64 * 2);
  bf16* A2   = (bf16*)carve((size_t)NPIX * K_C2 * 2);
  float* h2  = (float*)carve((size_t)NPIX * HDIM * 4);
  bf16* hb   = (bf16*)carve((size_t)NPIX * HDIM * 2);
  bf16* A3   = (bf16*)carve((size_t)NPIX * K_R1 * 2);
  float* r1  = (float*)carve((size_t)NPIX * RESH * 4);
  bf16* r1b  = (bf16*)carve((size_t)NPIX * RESH * 2);
  float* ze  = (float*)carve((size_t)NPIX * EDIM * 4);
  bf16* zeb  = (bf16*)carve((size_t)NPIX * EDIM * 2);
  float* dots= (float*)carve((size_t)NPIX * NE * 4);
  bf16* w2p  = (bf16*)carve((size_t)HDIM * K_C2 * 2);
  bf16* w1p  = (bf16*)carve((size_t)RESH * K_R1 * 2);
  bf16* w2b  = (bf16*)carve((size_t)HDIM * RESH * 2);
  bf16* pqb  = (bf16*)carve((size_t)EDIM * HDIM * 2);
  bf16* embb = (bf16*)carve((size_t)NE * EDIM * 2);
  float* en  = (float*)carve((size_t)NE * 4);
  int*  idxb = (int*)carve((size_t)NPIX * 4);
  int*  cnts = (int*)carve((size_t)NE * 4);
  float* mse = (float*)carve(256);
  (void)ws_size; (void)in_sizes; (void)n_in; (void)out_size;

  const int T = 256;
  auto blks = [](long n, int t) { return (int)((n + t - 1) / t); };
  auto gblk = [](int M, int N) { return (M / 128) * (N / 64); };

  // init histogram + mse accumulator
  k_init<<<blks(NE + 1, T), T, 0, stream>>>(cnts, mse);

  // encoder stage 1
  k_conv1<<<blks((long)NB * 128 * 128 * 64, T), T, 0, stream>>>(x, conv1_w, conv1_b, h1);

  // weight packs (OIHW -> [O][spatial*Ci] bf16)
  k_pack_w<<<blks(HDIM * K_C2, T), T, 0, stream>>>(conv2_w, w2p, HDIM, 64, 4, 4);
  k_pack_w<<<blks(RESH * K_R1, T), T, 0, stream>>>(res_w1, w1p, RESH, HDIM, 3, 3);
  k_pack_w<<<blks(HDIM * RESH, T), T, 0, stream>>>(res_w2, w2b, HDIM, RESH, 1, 1);
  k_pack_w<<<blks(EDIM * HDIM, T), T, 0, stream>>>(preq_w, pqb, EDIM, HDIM, 1, 1);
  k_pack_w<<<blks(NE * EDIM, T), T, 0, stream>>>(emb, embb, NE, EDIM, 1, 1);

  // conv2 as implicit GEMM (WMMA, double-buffered async-LDS B panel)
  k_im2col_c2<<<blks((long)NPIX * 16 * 8, T), T, 0, stream>>>(h1, A2);
  gemm_bf16_wmma<<<gblk(NPIX, HDIM), T, 0, stream>>>(
      A2, w2p, h2, conv2_b, NPIX, HDIM, K_C2, 0);

  // residual stack: 2 iterations, shared weights
  for (int it = 0; it < 2; ++it) {
    k_relu_cast<<<blks((long)NPIX * HDIM, T), T, 0, stream>>>(h2, hb, NPIX * HDIM, 1);
    k_im2col_r<<<blks((long)NPIX * 9 * 16, T), T, 0, stream>>>(hb, A3);
    gemm_bf16_wmma<<<gblk(NPIX, RESH), T, 0, stream>>>(
        A3, w1p, r1, nullptr, NPIX, RESH, K_R1, 0);
    k_relu_cast<<<blks((long)NPIX * RESH, T), T, 0, stream>>>(r1, r1b, NPIX * RESH, 1);
    gemm_bf16_wmma<<<gblk(NPIX, HDIM), T, 0, stream>>>(
        r1b, w2b, h2, nullptr, NPIX, HDIM, RESH, 1 /*accum*/);
  }

  // pre-quant 1x1
  k_relu_cast<<<blks((long)NPIX * HDIM, T), T, 0, stream>>>(h2, hb, NPIX * HDIM, 1);
  gemm_bf16_wmma<<<gblk(NPIX, EDIM), T, 0, stream>>>(
      hb, pqb, ze, preq_b, NPIX, EDIM, HDIM, 0);
  k_relu_cast<<<blks((long)NPIX * EDIM, T), T, 0, stream>>>(ze, zeb, NPIX * EDIM, 0);

  // VQ: dot products (WMMA) + argmin
  k_enorm<<<blks(NE, T), T, 0, stream>>>(emb, en);
  gemm_bf16_wmma<<<gblk(NPIX, NE), T, 0, stream>>>(
      zeb, embb, dots, nullptr, NPIX, NE, EDIM, 0);
  k_argmin<<<NPIX / 8, T, 0, stream>>>(dots, en, idxb, cnts);

  // outputs
  k_zq<<<blks((long)NPIX * EDIM, T), T, 0, stream>>>(idxb, emb, ze, out, mse);
  k_scalars<<<1, 1024, 0, stream>>>(cnts, mse, out);
  k_copyout<<<blks(NE * EDIM + NPIX, T), T, 0, stream>>>(emb, idxb, out);
}
